// CausalSelfAttention_72748156060118
// MI455X (gfx1250) — compile-verified, matrix operand order checked by reading, not run on hardware
//
#include <hip/hip_runtime.h>

// ---------------------------------------------------------------------------
// CDNA5 (gfx1250) fused causal self-attention + residual + LayerNorm.
// bf16 WMMA (v_wmma_f32_16x16x32_bf16) for all GEMMs, fp32 accumulation.
// K/V tiles staged to LDS via global_load_async_to_lds_b128 (ASYNCcnt).
// ---------------------------------------------------------------------------

typedef __attribute__((ext_vector_type(16))) __bf16         v16bf;
typedef __attribute__((ext_vector_type(8)))  float          v8f;
typedef __attribute__((ext_vector_type(8)))  unsigned short v8us;

#define NEG_INF_F (-1.0e9f)

__device__ __forceinline__ unsigned short f2bf(float f) {
    union { float f; unsigned u; } v; v.f = f;
    unsigned r = v.u + 0x7FFFu + ((v.u >> 16) & 1u);   // round-to-nearest-even
    return (unsigned short)(r >> 16);
}

union FragU { v16bf v; v8us h[2]; unsigned short u[16]; };

// Load a 16x32 bf16 fragment in the ISA "16-bit A 16x32" per-lane layout:
// lanes 0-15: row=lane,    K in {0..7} and {16..23}
// lanes16-31: row=lane-16, K in {8..15} and {24..31}
// Two 16-byte loads per lane. `base` points at (row0, k0); `stride` in halves.
__device__ __forceinline__ v16bf load_frag(const unsigned short* base, int stride) {
    int l = threadIdx.x & 31;
    const unsigned short* p = base + (size_t)(l & 15) * stride + ((l < 16) ? 0 : 8);
    FragU f;
    f.h[0] = *(const v8us*)(p);
    f.h[1] = *(const v8us*)(p + 16);
    return f.v;
}

__device__ __forceinline__ v8f wmma_bf16(v16bf a, v16bf b, v8f c) {
    return __builtin_amdgcn_wmma_f32_16x16x32_bf16(
        /*neg_a=*/false, a, /*neg_b=*/false, b,
        /*c_mod=*/(short)0, c, /*reuse_a=*/false, /*reuse_b=*/false);
}

// Keep A + all 8 B fragments simultaneously live so the scheduler cannot
// re-serialize load->wmma pairs: one s_wait per K-step, WMMAs back-to-back.
__device__ __forceinline__ void pin_frags(v16bf& a, v16bf (&b)[8]) {
    asm volatile("" : "+v"(a), "+v"(b[0]), "+v"(b[1]), "+v"(b[2]), "+v"(b[3]),
                      "+v"(b[4]), "+v"(b[5]), "+v"(b[6]), "+v"(b[7]));
}

// Async DMA of 16 bytes global -> LDS (per lane), tracked by ASYNCcnt.
// lds_off: workgroup-relative LDS byte offset (= low 32 bits of a generic
// shared pointer, per the CDNA5 LDS aperture rule).
__device__ __forceinline__ void async_ld_b128(unsigned lds_off, const void* gsrc) {
    unsigned long long ga = (unsigned long long)gsrc;
    asm volatile("global_load_async_to_lds_b128 %0, %1, off"
                 :: "v"(lds_off), "v"(ga) : "memory");
}

// ---------------------------------------------------------------------------
// Kernel 0: fp32 -> bf16 elementwise convert.
// ---------------------------------------------------------------------------
__global__ void cvt_bf16_kernel(const float* __restrict__ src,
                                unsigned short* __restrict__ dst, int n) {
    int i = blockIdx.x * blockDim.x + threadIdx.x;
    if (i < n) dst[i] = f2bf(src[i]);
}

// ---------------------------------------------------------------------------
// Kernel 1: convert + transpose all weight matrices (1024x1024 each) to bf16
// [N][K] layout so B-operand fragments load contiguously along K.
// blockIdx.y: 0=Wq 1=Wk 2=Wv 3=Wo.  dst[n*1024+k] = src[k*1024+n].
// ---------------------------------------------------------------------------
__global__ void cvt_wT_kernel(const float* __restrict__ Wq, const float* __restrict__ Wk,
                              const float* __restrict__ Wv, const float* __restrict__ Wo,
                              unsigned short* __restrict__ wqT, unsigned short* __restrict__ wkT,
                              unsigned short* __restrict__ wvT, unsigned short* __restrict__ woT) {
    int i = blockIdx.x * 256 + threadIdx.x;      // 0 .. 1048575
    int n = i & 1023, k = i >> 10;
    int z = blockIdx.y;
    const float* src = (z == 0) ? Wq : (z == 1) ? Wk : (z == 2) ? Wv : Wo;
    unsigned short* dst = (z == 0) ? wqT : (z == 1) ? wkT : (z == 2) ? wvT : woT;
    dst[(size_t)n * 1024 + k] = f2bf(src[(size_t)k * 1024 + n]);
}

// ---------------------------------------------------------------------------
// Kernel 2: QKV projection GEMM.  M=8192 (b,s), K=1024 (d), N=1024 (h,dk).
// Each wave computes a 16x128 strip (8 accumulators).  All 9 fragments of a
// K-step are loaded and pinned before the 8 WMMAs so LOADcnt waits collapse
// to one per K-step.  Bias folded in; 1/sqrt(dk)=0.125 folded into Q.
// Q,K stored [b,h][S][dk]; V stored transposed [b,h][dk][S].
// ---------------------------------------------------------------------------
__global__ void __launch_bounds__(256)
qkv_kernel(const unsigned short* __restrict__ xb,
           const unsigned short* __restrict__ wqT, const unsigned short* __restrict__ wkT,
           const unsigned short* __restrict__ wvT,
           const float* __restrict__ bq, const float* __restrict__ bk,
           const float* __restrict__ bv,
           unsigned short* __restrict__ qb, unsigned short* __restrict__ kb,
           unsigned short* __restrict__ vb) {
    const int S = 2048, D = 1024, H = 16, DK = 64;
    int wave = threadIdx.x >> 5, lane = threadIdx.x & 31;
    int m0    = blockIdx.x * 16;
    int nbase = wave * 128;
    int mat   = blockIdx.z;

    const unsigned short* W    = (mat == 0) ? wqT : (mat == 1) ? wkT : wvT;
    const float*          bias = (mat == 0) ? bq  : (mat == 1) ? bk  : bv;

    const unsigned short* Abase = xb + (size_t)m0 * D;
    const unsigned short* Bbase = W + (size_t)nbase * D;

    v8f acc[8] = {};
    for (int kk = 0; kk < D; kk += 32) {
        v16bf a = load_frag(Abase + kk, D);
        v16bf bfr[8];
#pragma unroll
        for (int t = 0; t < 8; ++t)
            bfr[t] = load_frag(Bbase + (size_t)(t * 16) * D + kk, D);
        __builtin_prefetch(Abase + kk + 256, 0, 1);   // global_prefetch_b8
        pin_frags(a, bfr);   // single wait, then 8 back-to-back WMMAs
#pragma unroll
        for (int t = 0; t < 8; ++t)
            acc[t] = wmma_bf16(a, bfr[t], acc[t]);
    }

    int hi = lane >> 4, nl = lane & 15;
    float scale = (mat == 0) ? 0.125f : 1.0f;
    unsigned short* dqk = (mat == 0) ? qb : kb;
#pragma unroll
    for (int t = 0; t < 8; ++t) {
        int nc = nbase + t * 16 + nl;
        float bval = bias[nc];
        int h = nc >> 6, dkk = nc & 63;
#pragma unroll
        for (int v = 0; v < 8; ++v) {
            int grow = m0 + v + 8 * hi;
            int b_ = grow >> 11, s = grow & 2047;
            unsigned short val = f2bf((acc[t][v] + bval) * scale);
            if (mat == 2)
                vb[(((size_t)b_ * H + h) * DK + dkk) * S + s] = val;
            else
                dqk[(((size_t)b_ * H + h) * S + s) * DK + dkk] = val;
        }
    }
}

// ---------------------------------------------------------------------------
// Kernel 3: flash attention.  One block per (b,h, 128-row query strip); each
// of the 8 waves owns a 16-row query tile.  All waves share the same kv walk,
// so each 32-kv block's K (32x64) and V^T (64x32) tiles are staged ONCE into
// LDS with global_load_async_to_lds_b128 (ASYNCcnt) and consumed by all
// waves.  Online softmax in C-layout; P staged bf16 per-wave through LDS.
// Compute is predicated per wave by its causal extent (wave-uniform branch,
// EXEC stays all-ones for WMMA); barriers run to the block-max extent.
// ---------------------------------------------------------------------------
struct AttnSmem {
    unsigned short Kt[32 * 64];        // 4 KB: rows = kv, contiguous dk
    unsigned short Vt[64 * 32];        // 4 KB: rows = dk, contiguous kv
    unsigned short Pt[8][16 * 32];     // 8 KB: per-wave P tile
};

__global__ void __launch_bounds__(256)
attn_kernel(const unsigned short* __restrict__ qb, const unsigned short* __restrict__ kb,
            const unsigned short* __restrict__ vb, unsigned short* __restrict__ ctxb) {
    const int S = 2048, DK = 64, HD = 1024;
    __shared__ __align__(16) AttnSmem sh;

    int tid = threadIdx.x;
    int wave = tid >> 5, lane = tid & 31;
    int qt = blockIdx.x * 8 + wave;       // 0..127
    int bh = blockIdx.y;                  // 0..63
    int q0 = qt * 16;

    const unsigned short* Qb = qb + ((size_t)bh * S + q0) * DK;
    const unsigned short* Kb = kb + (size_t)bh * S * DK;
    const unsigned short* Vb = vb + (size_t)bh * DK * S;

    v16bf qf0 = load_frag(Qb, DK);        // K = 0..31  of dk
    v16bf qf1 = load_frag(Qb + 32, DK);   // K = 32..63 of dk

    v8f o0 = {}, o1 = {}, o2 = {}, o3 = {};
    float m[8], lsum[8];
#pragma unroll
    for (int v = 0; v < 8; ++v) { m[v] = -3.0e38f; lsum[v] = 0.f; }

    int hi = lane >> 4, nl = lane & 15;
    unsigned short* Pw = &sh.Pt[wave][0];

    int nb    = (q0 + 47) >> 5;                              // this wave's kv blocks
    int nbmax = ((int)(blockIdx.x * 8 + 7) * 16 + 47) >> 5;  // block max

    // Cooperative staging map: 256 threads x 16B = one 4KB tile per issue.
    int krow = tid >> 3, kch = tid & 7;     // K tile: 32 rows x 128B
    int vrow = tid >> 2, vch = tid & 3;     // V tile: 64 rows x 64B
    unsigned kdst = (unsigned)(size_t)&sh.Kt[krow * 64 + kch * 8];
    unsigned vdst = (unsigned)(size_t)&sh.Vt[vrow * 32 + vch * 8];

    for (int blk = 0; blk < nbmax; ++blk) {
        int c0 = blk * 32;
        __syncthreads();   // previous tile fully consumed
        async_ld_b128(kdst, Kb + (size_t)(c0 + krow) * DK + kch * 8);
        async_ld_b128(vdst, Vb + (size_t)vrow * S + c0 + vch * 8);
        asm volatile("s_wait_asynccnt 0" ::: "memory");
        __syncthreads();   // tile visible to all waves

        if (blk < nb) {
            // ---- scores S = Q*K^T over this 16x32 kv block (2 n-tiles) ----
            v8f s0 = {}, s1 = {};
            {
                v16bf b00 = load_frag(sh.Kt, DK);
                v16bf b10 = load_frag(sh.Kt + 16 * DK, DK);
                v16bf b01 = load_frag(sh.Kt + 32, DK);
                v16bf b11 = load_frag(sh.Kt + 16 * DK + 32, DK);
                s0 = wmma_bf16(qf0, b00, s0);
                s1 = wmma_bf16(qf0, b10, s1);
                s0 = wmma_bf16(qf1, b01, s0);
                s1 = wmma_bf16(qf1, b11, s1);
            }
            bool need_mask = (c0 + 31 > q0);
            // ---- causal mask + online softmax (per-row half-wave shuffles) ----
#pragma unroll
            for (int v = 0; v < 8; ++v) {
                int grow = q0 + v + 8 * hi;
                if (need_mask) {
                    if (c0 + nl > grow)      s0[v] = NEG_INF_F;
                    if (c0 + 16 + nl > grow) s1[v] = NEG_INF_F;
                }
                float mx = fmaxf(s0[v], s1[v]);
                mx = fmaxf(mx, __shfl_xor(mx, 1));
                mx = fmaxf(mx, __shfl_xor(mx, 2));
                mx = fmaxf(mx, __shfl_xor(mx, 4));
                mx = fmaxf(mx, __shfl_xor(mx, 8));
                float mn    = fmaxf(m[v], mx);
                float alpha = __expf(m[v] - mn);
                float p0 = __expf(s0[v] - mn);
                float p1 = __expf(s1[v] - mn);
                float rs = p0 + p1;
                rs += __shfl_xor(rs, 1); rs += __shfl_xor(rs, 2);
                rs += __shfl_xor(rs, 4); rs += __shfl_xor(rs, 8);
                lsum[v] = lsum[v] * alpha + rs;
                m[v] = mn;
                o0[v] *= alpha; o1[v] *= alpha; o2[v] *= alpha; o3[v] *= alpha;
                int r = v + 8 * hi;
                Pw[r * 32 + nl]      = f2bf(p0);
                Pw[r * 32 + 16 + nl] = f2bf(p1);
            }
            asm volatile("s_wait_dscnt 0" ::: "memory");  // P visible (same wave)
            // ---- ctx += P * V  (V^T rows contiguous along kv, from LDS) ----
            v16bf pf  = load_frag(Pw, 32);
            v16bf v0f = load_frag(sh.Vt + 0 * 16 * 32, 32);
            v16bf v1f = load_frag(sh.Vt + 1 * 16 * 32, 32);
            v16bf v2f = load_frag(sh.Vt + 2 * 16 * 32, 32);
            v16bf v3f = load_frag(sh.Vt + 3 * 16 * 32, 32);
            o0 = wmma_bf16(pf, v0f, o0);
            o1 = wmma_bf16(pf, v1f, o1);
            o2 = wmma_bf16(pf, v2f, o2);
            o3 = wmma_bf16(pf, v3f, o3);
        }
    }

    int b_ = bh >> 4, h = bh & 15;
#pragma unroll
    for (int v = 0; v < 8; ++v) {
        float inv = 1.0f / lsum[v];
        int grow = q0 + v + 8 * hi;
        size_t row = ((size_t)b_ * S + grow) * HD + (size_t)h * DK;
        ctxb[row +  0 + nl] = f2bf(o0[v] * inv);
        ctxb[row + 16 + nl] = f2bf(o1[v] * inv);
        ctxb[row + 32 + nl] = f2bf(o2[v] * inv);
        ctxb[row + 48 + nl] = f2bf(o3[v] * inv);
    }
}

// ---------------------------------------------------------------------------
// Kernel 4: out = LayerNorm(x + ctx*Wo + bo).  Each block owns 16 full rows
// (16 x 1024): wave w computes cols [128w, 128w+128) via 8 WMMA accumulators
// (fragments batched + pinned per K-step like qkv), then block-wide LDS
// reduction for mean/var, then normalize + store fp32.
// ---------------------------------------------------------------------------
__global__ void __launch_bounds__(256)
outln_kernel(const unsigned short* __restrict__ ctxb, const unsigned short* __restrict__ woT,
             const float* __restrict__ bo, const float* __restrict__ x,
             const float* __restrict__ gamma, const float* __restrict__ beta,
             float* __restrict__ out) {
    const int D = 1024, HD = 1024;
    __shared__ float sums[16], ssq[16];

    int wave = threadIdx.x >> 5, lane = threadIdx.x & 31;
    int m0 = blockIdx.x * 16;
    if (threadIdx.x < 16) { sums[threadIdx.x] = 0.f; ssq[threadIdx.x] = 0.f; }
    __syncthreads();

    const unsigned short* Abase = ctxb + (size_t)m0 * HD;
    const unsigned short* Bbase = woT + (size_t)(wave * 128) * HD;
    int nbase = wave * 128;

    v8f acc[8] = {};
    for (int kk = 0; kk < HD; kk += 32) {
        v16bf a = load_frag(Abase + kk, HD);
        v16bf bfr[8];
#pragma unroll
        for (int t = 0; t < 8; ++t)
            bfr[t] = load_frag(Bbase + (size_t)(t * 16) * HD + kk, HD);
        __builtin_prefetch(Abase + kk + 256, 0, 1);
        pin_frags(a, bfr);
#pragma unroll
        for (int t = 0; t < 8; ++t)
            acc[t] = wmma_bf16(a, bfr[t], acc[t]);
    }

    int hi = lane >> 4, nl = lane & 15;
    float psum[8], psq[8];
#pragma unroll
    for (int v = 0; v < 8; ++v) { psum[v] = 0.f; psq[v] = 0.f; }
#pragma unroll
    for (int t = 0; t < 8; ++t) {
        int n = nbase + t * 16 + nl;
        float bv = bo[n];
#pragma unroll
        for (int v = 0; v < 8; ++v) {
            int grow = m0 + v + 8 * hi;
            float y = acc[t][v] + bv + x[(size_t)grow * D + n];
            acc[t][v] = y;
            psum[v] += y;
            psq[v]  += y * y;
        }
    }
#pragma unroll
    for (int v = 0; v < 8; ++v) {
        float s = psum[v], q = psq[v];
        s += __shfl_xor(s, 1); s += __shfl_xor(s, 2);
        s += __shfl_xor(s, 4); s += __shfl_xor(s, 8);
        q += __shfl_xor(q, 1); q += __shfl_xor(q, 2);
        q += __shfl_xor(q, 4); q += __shfl_xor(q, 8);
        if (nl == 0) {                       // lanes 0 and 16 hold row totals
            atomicAdd(&sums[v + 8 * hi], s); // ds_add_f32
            atomicAdd(&ssq[v + 8 * hi],  q);
        }
    }
    __syncthreads();
#pragma unroll
    for (int v = 0; v < 8; ++v) {
        int r = v + 8 * hi;
        float mu   = sums[r] * (1.0f / 1024.0f);
        float var  = ssq[r] * (1.0f / 1024.0f) - mu * mu;
        float rstd = rsqrtf(var + 1e-6f);
        int grow = m0 + r;
#pragma unroll
        for (int t = 0; t < 8; ++t) {
            int n = nbase + t * 16 + nl;
            out[(size_t)grow * D + n] = (acc[t][v] - mu) * rstd * gamma[n] + beta[n];
        }
    }
}

// ---------------------------------------------------------------------------
// Host launcher.
// ---------------------------------------------------------------------------
extern "C" void kernel_launch(void* const* d_in, const int* in_sizes, int n_in,
                              void* d_out, int out_size, void* d_ws, size_t ws_size,
                              hipStream_t stream) {
    const float* x     = (const float*)d_in[0];
    const float* Wq    = (const float*)d_in[1];
    const float* bq    = (const float*)d_in[2];
    const float* Wk    = (const float*)d_in[3];
    const float* bk    = (const float*)d_in[4];
    const float* Wv    = (const float*)d_in[5];
    const float* bv    = (const float*)d_in[6];
    const float* Wo    = (const float*)d_in[7];
    const float* bo    = (const float*)d_in[8];
    const float* gamma = (const float*)d_in[9];
    const float* beta  = (const float*)d_in[10];
    float* out = (float*)d_out;
    (void)in_sizes; (void)n_in; (void)out_size; (void)ws_size;

    const size_t BS = 8192, D = 1024, HD = 1024, BH = 64, S = 2048, DK = 64;
    char* w = (char*)d_ws;
    size_t off = 0;
    auto take = [&](size_t bytes) -> unsigned short* {
        unsigned short* p = (unsigned short*)(w + off);
        off += (bytes + 255) & ~(size_t)255;
        return p;
    };
    unsigned short* xb   = take(BS * D * 2);       // x bf16
    unsigned short* wqT  = take(D * HD * 2);       // Wq^T bf16 [N][K]
    unsigned short* wkT  = take(D * HD * 2);
    unsigned short* wvT  = take(D * HD * 2);
    unsigned short* woT  = take(HD * D * 2);       // Wo^T bf16 [d][hk]
    unsigned short* qbuf = take(BH * S * DK * 2);  // [b,h][S][dk]
    unsigned short* kbuf = take(BH * S * DK * 2);  // [b,h][S][dk]
    unsigned short* vbuf = take(BH * DK * S * 2);  // [b,h][dk][S]  (transposed)
    unsigned short* ctxb = take(BS * HD * 2);      // [b*s][h*dk] bf16

    cvt_bf16_kernel<<<(int)(BS * D / 256), 256, 0, stream>>>(x, xb, (int)(BS * D));
    cvt_wT_kernel<<<dim3(4096, 4), 256, 0, stream>>>(Wq, Wk, Wv, Wo, wqT, wkT, wvT, woT);
    qkv_kernel<<<dim3(512, 1, 3), 256, 0, stream>>>(xb, wqT, wkT, wvT,
                                                    bq, bk, bv, qbuf, kbuf, vbuf);
    attn_kernel<<<dim3(16, 64), 256, 0, stream>>>(qbuf, kbuf, vbuf, ctxb);
    outln_kernel<<<dim3(512), 256, 0, stream>>>(ctxb, woT, bo, x, gamma, beta, out);
}
